// BlockAttentionProduct_72121090834678
// MI455X (gfx1250) — matrix operand order repaired and checked by reading, not compiled, not run on hardware
//
#include <hip/hip_runtime.h>

#define B_    4
#define H_    16
#define T_    4096
#define D_    64
#define G_    128
#define BLK_  128
#define NB_   (T_ / BLK_)     // 32 sequence blocks
#define NK_   (BLK_ + G_)     // 256 keys per block (local + global)

// padded LDS row strides (in elements) to rotate bank mapping per row
#define QH_LD 72              // Qh/Vh: 144 B rows
#define VH_LD 72
#define KT_LD 264             // Kt: 528 B rows (16B aligned)
#define S_LD  264             // S/P (f16): 528 B rows (16B aligned)

typedef __attribute__((ext_vector_type(16))) _Float16 v16h;
typedef __attribute__((ext_vector_type(4)))  _Float16 v4h;
typedef __attribute__((ext_vector_type(8)))  float    v8f;

// A-fragment (16 x 32 slab of f16, row-major, row stride ld): lane -> row
// (lane&15); regs -> K: lanes 0-15 hold K={0..7,16..23}+k0, lanes 16-31 hold
// K={8..15,24..31}+k0 (ISA 7.12.2, 16-bit A 16x32). Two 16B contiguous runs.
__device__ __forceinline__ v16h frag_A(const _Float16* __restrict__ base, int ld,
                                       int k0, int lane) {
  const _Float16* p = base + (lane & 15) * ld + k0 + ((lane >> 4) << 3);
  v16h a;
#pragma unroll
  for (int e = 0; e < 8; ++e) a[e] = p[e];
#pragma unroll
  for (int e = 0; e < 8; ++e) a[8 + e] = p[16 + e];
  return a;
}

__global__ __launch_bounds__(256) void block_attn_wmma_kernel(
    const float* __restrict__ q, const float* __restrict__ k,
    const float* __restrict__ v, const float* __restrict__ amask,
    const float* __restrict__ gk, const float* __restrict__ gv,
    const float* __restrict__ gmask, float* __restrict__ out) {
  extern __shared__ float4 smemraw[];
  _Float16* Qh  = (_Float16*)smemraw;            // 128 x QH_LD f16   (18 KB)
  _Float16* Kt  = Qh + BLK_ * QH_LD;             // 64 x KT_LD f16    (33 KB)
  _Float16* Vh  = Kt + D_ * KT_LD;               // 256 x VH_LD f16   (36 KB)
  _Float16* S   = Vh + NK_ * VH_LD;              // 128 x S_LD f16    (66 KB)
  float*    red = (float*)(S + BLK_ * S_LD);     // 256 f32 (softmax partials)

  const int bh   = blockIdx.x / NB_;
  const int nblk = blockIdx.x % NB_;
  const int tid  = threadIdx.x;
  const int lane = tid & 31;
  const int wave = tid >> 5;

  const float4* Qp4 = (const float4*)(q + (size_t)(bh * T_ + nblk * BLK_) * D_);
  const float4* Kp4 = (const float4*)(k + (size_t)(bh * T_ + nblk * BLK_) * D_);
  const float4* Vp4 = (const float4*)(v + (size_t)(bh * T_ + nblk * BLK_) * D_);
  const float4* GK4 = (const float4*)(gk + (size_t)bh * G_ * D_);
  const float4* GV4 = (const float4*)(gv + (size_t)bh * G_ * D_);
  const float*  AM  = amask + (size_t)bh * T_ + nblk * BLK_;
  const float*  GM  = gmask + (size_t)bh * G_;

  // ---- Stage 1: b128 global loads -> f16 LDS (K stored transposed) ----
  for (int i = tid; i < (BLK_ * D_) / 4; i += 256) {  // 8 iterations
    const int key = (4 * i) / D_;   // 0..127
    const int dep = (4 * i) % D_;   // multiple of 4
    const float4 xq = Qp4[i];
    *(v4h*)&Qh[key * QH_LD + dep] =
        (v4h){(_Float16)xq.x, (_Float16)xq.y, (_Float16)xq.z, (_Float16)xq.w};
    const float4 xv = Vp4[i];
    *(v4h*)&Vh[key * VH_LD + dep] =
        (v4h){(_Float16)xv.x, (_Float16)xv.y, (_Float16)xv.z, (_Float16)xv.w};
    const float4 xgv = GV4[i];
    *(v4h*)&Vh[(key + BLK_) * VH_LD + dep] =
        (v4h){(_Float16)xgv.x, (_Float16)xgv.y, (_Float16)xgv.z, (_Float16)xgv.w};
    const float4 xk = Kp4[i];
    Kt[(dep + 0) * KT_LD + key] = (_Float16)xk.x;
    Kt[(dep + 1) * KT_LD + key] = (_Float16)xk.y;
    Kt[(dep + 2) * KT_LD + key] = (_Float16)xk.z;
    Kt[(dep + 3) * KT_LD + key] = (_Float16)xk.w;
    const float4 xgk = GK4[i];
    Kt[(dep + 0) * KT_LD + BLK_ + key] = (_Float16)xgk.x;
    Kt[(dep + 1) * KT_LD + BLK_ + key] = (_Float16)xgk.y;
    Kt[(dep + 2) * KT_LD + BLK_ + key] = (_Float16)xgk.z;
    Kt[(dep + 3) * KT_LD + BLK_ + key] = (_Float16)xgk.w;
  }
  __syncthreads();

  // ---- Stage 2: S = f16((Q.K'^T)/sqrt(d) + mask); wave w -> rows [16w,16w+16) ----
  {
    const int mBase  = wave * 16;
    const v16h a0 = frag_A(Qh + mBase * QH_LD, QH_LD, 0, lane);
    const v16h a1 = frag_A(Qh + mBase * QH_LD, QH_LD, 32, lane);
    const int rowOff = mBase + ((lane >> 4) << 3);
    const int col    = lane & 15;
#pragma unroll
    for (int t = 0; t < NK_ / 16; ++t) {
      // B[k][n] = K'[n][k] = Kt[k][n]: lane -> key n=(lane&15), regs -> depth k (same order as A)
      v16h b0, b1;
#pragma unroll
      for (int e = 0; e < 8; ++e) {
        const int kk = ((lane >> 4) << 3) + e;
        b0[e]     = Kt[kk * KT_LD + t * 16 + (lane & 15)];
        b0[8 + e] = Kt[(16 + kk) * KT_LD + t * 16 + (lane & 15)];
        b1[e]     = Kt[(32 + kk) * KT_LD + t * 16 + (lane & 15)];
        b1[8 + e] = Kt[(48 + kk) * KT_LD + t * 16 + (lane & 15)];
      }
      v8f c = {};
      c = __builtin_amdgcn_wmma_f32_16x16x32_f16(false, a0, false, b0,
                                                 (short)0, c, false, false);
      c = __builtin_amdgcn_wmma_f32_16x16x32_f16(false, a1, false, b1,
                                                 (short)0, c, false, false);
      const float maskv = (t < 8) ? AM[t * 16 + col] : GM[t * 16 + col - BLK_];
#pragma unroll
      for (int r = 0; r < 8; ++r)
        S[(rowOff + r) * S_LD + t * 16 + col] =
            (_Float16)(c[r] * 0.125f + maskv);
    }
  }
  __syncthreads();

  // ---- Stage 3: softmax (f32 math on f16 rows), 2 threads per row; P := S ----
  {
    const int row  = tid & (BLK_ - 1);
    const int half = tid >> 7;          // 0 or 1
    _Float16* Srow = S + row * S_LD + half * (NK_ / 2);
    float mx = -3.4e38f;
#pragma unroll 4
    for (int c = 0; c < NK_ / 2; ++c) mx = fmaxf(mx, (float)Srow[c]);
    red[tid] = mx;
    __syncthreads();
    const float m = fmaxf(red[row], red[row + BLK_]);
    float sum = 0.f;
#pragma unroll 4
    for (int c = 0; c < NK_ / 2; ++c) {
      const float e = __expf((float)Srow[c] - m);
      Srow[c] = (_Float16)e;
      sum += e;
    }
    __syncthreads();                    // all reads of red (maxes) done
    red[tid] = sum;
    __syncthreads();
    const float inv = 1.f / (red[row] + red[row + BLK_]);
#pragma unroll 4
    for (int c = 0; c < NK_ / 2; ++c)
      Srow[c] = (_Float16)((float)Srow[c] * inv);   // in place, same type
  }
  __syncthreads();

  // ---- Stage 4: O = P . V'  (P aliases S) ----
  {
    const int mBase  = wave * 16;
    const int rowOff = mBase + ((lane >> 4) << 3);
    const int col    = lane & 15;
    v8f acc[4] = {};
#pragma unroll
    for (int kt = 0; kt < NK_ / 32; ++kt) {
      const v16h a = frag_A(S + mBase * S_LD, S_LD, kt * 32, lane);
#pragma unroll
      for (int nt = 0; nt < 4; ++nt) {
        // B[k][n] = V'[k][n]: lane -> d-column n=(lane&15), regs -> key k (same order as A)
        v16h b;
#pragma unroll
        for (int e = 0; e < 8; ++e) {
          const int kk = kt * 32 + ((lane >> 4) << 3) + e;
          b[e]     = Vh[kk * VH_LD + nt * 16 + (lane & 15)];
          b[8 + e] = Vh[(16 + kk) * VH_LD + nt * 16 + (lane & 15)];
        }
        acc[nt] = __builtin_amdgcn_wmma_f32_16x16x32_f16(false, a, false, b,
                                                         (short)0, acc[nt],
                                                         false, false);
      }
    }
    float* Op = out + (size_t)(bh * T_ + nblk * BLK_) * D_;
#pragma unroll
    for (int nt = 0; nt < 4; ++nt)
#pragma unroll
      for (int r = 0; r < 8; ++r)
        Op[(rowOff + r) * D_ + nt * 16 + col] = acc[nt][r];
  }
}

extern "C" void kernel_launch(void* const* d_in, const int* in_sizes, int n_in,
                              void* d_out, int out_size, void* d_ws,
                              size_t ws_size, hipStream_t stream) {
  (void)in_sizes; (void)n_in; (void)out_size; (void)d_ws; (void)ws_size;
  const float* q  = (const float*)d_in[0];
  const float* k  = (const float*)d_in[1];
  const float* v  = (const float*)d_in[2];
  const float* am = (const float*)d_in[3];
  const float* gk = (const float*)d_in[4];
  const float* gv = (const float*)d_in[5];
  const float* gm = (const float*)d_in[6];
  float* out = (float*)d_out;

  const size_t lds = (size_t)(BLK_ * QH_LD) * 2     // Qh
                   + (size_t)(D_ * KT_LD) * 2       // Kt
                   + (size_t)(NK_ * VH_LD) * 2      // Vh
                   + (size_t)(BLK_ * S_LD) * 2      // S (aliased as P)
                   + 256 * 4;                       // red  => ~154 KB
                                                    // -> 2 workgroups per WGP

  dim3 grid(B_ * H_ * NB_);  // 2048 workgroups
  block_attn_wmma_kernel<<<grid, 256, lds, stream>>>(q, k, v, am, gk, gv, gm,
                                                     out);
}
